// KMeansFSQ_32315333935397
// MI455X (gfx1250) — matrix operand order repaired and checked by reading, not compiled, not run on hardware
//
#include <hip/hip_runtime.h>

#define NCLUST 1024
#define EMBED  512
#define NTOK   65536
#define DECAYF 0.99f
#define ODECAY 0.01f
#define EPSV   1e-5f
#define COMMITF 0.25f

typedef __attribute__((ext_vector_type(16))) __bf16      v16bf;
typedef __attribute__((ext_vector_type(8)))  float       v8f;
typedef __attribute__((ext_vector_type(8)))  unsigned int v8u;
typedef __attribute__((ext_vector_type(4)))  unsigned int v4u;

// ---- workspace layout (float offsets) ----
#define WS_SUM    ((size_t)0)
#define WS_SUMSQ  ((size_t)512)
#define WS_NMEAN  ((size_t)1024)
#define WS_NSTD   ((size_t)1536)
#define WS_NINV   ((size_t)2048)
#define WS_CNORM  ((size_t)2560)
#define WS_USAGE  ((size_t)3584)
#define WS_LOSS   ((size_t)4608)
#define WS_CSUM   ((size_t)5120)
#define WS_ZERO_END (WS_CSUM + (size_t)NCLUST*EMBED)      // 529408 floats
#define WS_CB_BYTE_OFF (WS_ZERO_END * 4)                  // 2,117,632 B (32B aligned)

// ---- output layout (float offsets, concatenated return tuple) ----
#define OUT_Q      ((size_t)0)
#define OUT_IDX    ((size_t)NTOK*EMBED)                   // 33554432
#define OUT_LOSS   (OUT_IDX + NTOK)                       // 33619968
#define OUT_NCB    (OUT_LOSS + 1)                         // 33619969 (odd -> scalar stores)
#define OUT_NUSAGE (OUT_NCB + (size_t)NCLUST*EMBED)
#define OUT_NCSUM  (OUT_NUSAGE + NCLUST)
#define OUT_NMEANS (OUT_NCSUM + (size_t)NCLUST*EMBED)
#define OUT_NSTDS  (OUT_NMEANS + EMBED)

__device__ __forceinline__ unsigned short f2bf(float f) {
  unsigned int u = __float_as_uint(f);
  unsigned int r = u + 0x7FFFu + ((u >> 16) & 1u);   // RNE
  return (unsigned short)(r >> 16);
}

__global__ void k_zero(float* wsf, int n) {
  int i = blockIdx.x * 256 + threadIdx.x;
  if (i < n) wsf[i] = 0.0f;
}

// per-channel sum / sumsq; block handles 512 rows, thread = channel (coalesced)
__global__ void k_stats(const float* __restrict__ x, float* wsf) {
  int c  = threadIdx.x;            // 512 threads
  int r0 = blockIdx.x * 512;
  float s1 = 0.f, s2 = 0.f;
  for (int r = 0; r < 512; ++r) {
    float v = x[(size_t)(r0 + r) * EMBED + c];
    s1 += v; s2 += v * v;
  }
  atomicAdd(&wsf[WS_SUM + c],   s1);
  atomicAdd(&wsf[WS_SUMSQ + c], s2);
}

__global__ void k_finstats(const float* __restrict__ cm, const float* __restrict__ cs,
                           float* wsf, float* out) {
  int c = threadIdx.x;             // 512 threads
  float s1 = wsf[WS_SUM + c], s2 = wsf[WS_SUMSQ + c];
  float m   = s1 * (1.0f / (float)NTOK);
  float var = (s2 - (float)NTOK * m * m) * (1.0f / (float)(NTOK - 1));
  var = var > 0.f ? var : 0.f;
  float s  = sqrtf(var) + EPSV;
  float nm = cm[c] * DECAYF + m * ODECAY;
  float ns = cs[c] * DECAYF + s * ODECAY;
  wsf[WS_NMEAN + c] = nm;
  wsf[WS_NSTD  + c] = ns;
  wsf[WS_NINV  + c] = 1.0f / ns;
  out[OUT_NMEANS + c] = nm;
  out[OUT_NSTDS  + c] = ns;
}

// codebook -> bf16 (workspace) + 0.5*||c_j||^2
__global__ void k_cbprep(const float* __restrict__ cb, float* wsf, unsigned short* cb16) {
  int j = blockIdx.x, t = threadIdx.x;   // 128 threads
  float4 v = *(const float4*)(cb + (size_t)j * EMBED + t * 4);
  unsigned int p0 = (unsigned int)f2bf(v.x) | ((unsigned int)f2bf(v.y) << 16);
  unsigned int p1 = (unsigned int)f2bf(v.z) | ((unsigned int)f2bf(v.w) << 16);
  *(uint2*)(cb16 + (size_t)j * EMBED + t * 4) = make_uint2(p0, p1);
  __shared__ float red[128];
  red[t] = v.x*v.x + v.y*v.y + v.z*v.z + v.w*v.w;
  __syncthreads();
  for (int s = 64; s > 0; s >>= 1) { if (t < s) red[t] += red[t + s]; __syncthreads(); }
  if (t == 0) wsf[WS_CNORM + j] = 0.5f * red[0];
}

// main: 64 rows/block, 4 waves x 16 rows, full 1024-cluster argmin via bf16 WMMA
__global__ void __launch_bounds__(128) k_main(const float* __restrict__ x,
                                              const float* __restrict__ cbf32,
                                              const unsigned short* __restrict__ cb16,
                                              float* wsf, float* out) {
  __shared__ unsigned short xs[64][EMBED + 8];   // normalized bf16 rows, identity k layout
  __shared__ int   sIdx[64];
  __shared__ float sred[128];

  const int t = threadIdx.x;
  const int lane = t & 31, wave = t >> 5;
  const size_t base = (size_t)blockIdx.x * 64;

  // ---- phase 1: load + normalize 64 rows into LDS (bf16, identity k layout) ----
  const int c0 = t * 4;                              // 0..508
  float4 mn = *(const float4*)(wsf + WS_NMEAN + c0);
  float4 iv = *(const float4*)(wsf + WS_NINV  + c0);
  // identity position within each 32-k group (contiguous k, matches B)
  const int g = c0 >> 5, sub = (c0 >> 3) & 3;
  const int pos = g * 32 + sub * 8 + (c0 & 7);
  for (int r = 0; r < 64; ++r) {
    float4 v = *(const float4*)(x + (base + r) * EMBED + c0);
    unsigned int p0 = (unsigned int)f2bf((v.x - mn.x) * iv.x) |
                      ((unsigned int)f2bf((v.y - mn.y) * iv.y) << 16);
    unsigned int p1 = (unsigned int)f2bf((v.z - mn.z) * iv.z) |
                      ((unsigned int)f2bf((v.w - mn.w) * iv.w) << 16);
    *(uint2*)&xs[r][pos] = make_uint2(p0, p1);
  }
  __syncthreads();

  // ---- phase 2: WMMA distance sweep with running argmin ----
  const float* cnorm = wsf + WS_CNORM;
  const int nlo = lane & 15, hi = lane >> 4;
  const int arow = wave * 16 + nlo;                  // A row for this lane
  float bval[8]; int bidx[8];
  #pragma unroll
  for (int r = 0; r < 8; ++r) { bval[r] = 3.4e38f; bidx[r] = 0; }

  for (int ct = 0; ct < 32; ++ct) {                  // 2 column tiles (32 clusters) per iter
    v8f acc0 = {0.f,0.f,0.f,0.f,0.f,0.f,0.f,0.f};
    v8f acc1 = acc0;
    const unsigned short* b0p = cb16 + (size_t)(ct * 32 + nlo) * EMBED + hi * 16;
    const unsigned short* b1p = b0p + (size_t)16 * EMBED;
    #pragma unroll
    for (int ks = 0; ks < 16; ++ks) {
      // A fragment: 16 contiguous bf16 (contiguous k layout) -> 2x b128 LDS loads
      const unsigned short* ap = &xs[arow][ks * 32 + hi * 16];
      v4u alo = *(const v4u*)ap;
      v4u ahi = *(const v4u*)(ap + 8);
      v8u au; au[0]=alo[0]; au[1]=alo[1]; au[2]=alo[2]; au[3]=alo[3];
              au[4]=ahi[0]; au[5]=ahi[1]; au[6]=ahi[2]; au[7]=ahi[3];
      v16bf a = __builtin_bit_cast(v16bf, au);
      // B fragments: 16 contiguous bf16 from bf16 codebook (L2/WGP$-resident)
      v4u b0l = *(const v4u*)(b0p + ks * 32);
      v4u b0h = *(const v4u*)(b0p + ks * 32 + 8);
      v8u bu0; bu0[0]=b0l[0]; bu0[1]=b0l[1]; bu0[2]=b0l[2]; bu0[3]=b0l[3];
               bu0[4]=b0h[0]; bu0[5]=b0h[1]; bu0[6]=b0h[2]; bu0[7]=b0h[3];
      v16bf b0 = __builtin_bit_cast(v16bf, bu0);
      v4u b1l = *(const v4u*)(b1p + ks * 32);
      v4u b1h = *(const v4u*)(b1p + ks * 32 + 8);
      v8u bu1; bu1[0]=b1l[0]; bu1[1]=b1l[1]; bu1[2]=b1l[2]; bu1[3]=b1l[3];
               bu1[4]=b1h[0]; bu1[5]=b1h[1]; bu1[6]=b1h[2]; bu1[7]=b1h[3];
      v16bf b1 = __builtin_bit_cast(v16bf, bu1);
      acc0 = __builtin_amdgcn_wmma_f32_16x16x32_bf16(false, a, false, b0, (short)0, acc0, false, false);
      acc1 = __builtin_amdgcn_wmma_f32_16x16x32_bf16(false, a, false, b1, (short)0, acc1, false, false);
    }
    const int n0 = ct * 32 + nlo, n1 = n0 + 16;
    const float cn0 = cnorm[n0], cn1 = cnorm[n1];
    #pragma unroll
    for (int r = 0; r < 8; ++r) {
      float s0 = cn0 - acc0[r];
      if (s0 < bval[r]) { bval[r] = s0; bidx[r] = n0; }
      float s1 = cn1 - acc1[r];
      if (s1 < bval[r]) { bval[r] = s1; bidx[r] = n1; }
    }
  }

  // ---- argmin across the 16 column-lanes (C/D layout: rows 0-7 lanes 0-15, rows 8-15 lanes 16-31)
  #pragma unroll
  for (int r = 0; r < 8; ++r) {
    float v = bval[r]; int id = bidx[r];
    #pragma unroll
    for (int off = 8; off >= 1; off >>= 1) {
      float ov = __shfl_xor(v, off, 16);
      int  oid = __shfl_xor(id, off, 16);
      if (ov < v || (ov == v && oid < id)) { v = ov; id = oid; }
    }
    if (nlo == 0) sIdx[wave * 16 + r + hi * 8] = id;
  }
  __syncthreads();

  // ---- phase 3: quantized output, indices, segment sums, loss ----
  float4 ns4 = *(const float4*)(wsf + WS_NSTD + c0);
  float lossAcc = 0.f;
  for (int r = 0; r < 64; ++r) {
    int idx = sIdx[r];
    size_t row = base + r;
    float4 xv = *(const float4*)(x     + row * EMBED + c0);
    float4 cv = *(const float4*)(cbf32 + (size_t)idx * EMBED + c0);
    float4 q;
    q.x = cv.x * ns4.x + mn.x; q.y = cv.y * ns4.y + mn.y;
    q.z = cv.z * ns4.z + mn.z; q.w = cv.w * ns4.w + mn.w;
    *(float4*)(out + OUT_Q + row * EMBED + c0) = q;
    float dx = xv.x - q.x, dy = xv.y - q.y, dz = xv.z - q.z, dw = xv.w - q.w;
    lossAcc += dx*dx + dy*dy + dz*dz + dw*dw;
    float* csp = wsf + WS_CSUM + (size_t)idx * EMBED + c0;
    atomicAdd(csp + 0, xv.x); atomicAdd(csp + 1, xv.y);
    atomicAdd(csp + 2, xv.z); atomicAdd(csp + 3, xv.w);
  }
  if (t < 64) {
    int idx = sIdx[t];
    atomicAdd(wsf + WS_USAGE + idx, 1.0f);
    out[OUT_IDX + base + t] = (float)idx;
  }
  sred[t] = lossAcc;
  __syncthreads();
  for (int s = 64; s > 0; s >>= 1) { if (t < s) sred[t] += sred[t + s]; __syncthreads(); }
  if (t == 0) atomicAdd(wsf + WS_LOSS, sred[0]);
}

// EMA update of usage/csum, new codebook, final loss
__global__ void k_final(const float* __restrict__ cu, const float* __restrict__ csum_in,
                        const float* __restrict__ wsf, float* out) {
  int j = blockIdx.x, t = threadIdx.x;   // 128 threads
  float nu = cu[j] * DECAYF + wsf[WS_USAGE + j] * ODECAY;
  if (t == 0) out[OUT_NUSAGE + j] = nu;
  float inv = 1.0f / (nu + EPSV);
  int c = t * 4;
  float4 old = *(const float4*)(csum_in + (size_t)j * EMBED + c);
  float4 acc = *(const float4*)(wsf + WS_CSUM + (size_t)j * EMBED + c);
  float n0 = old.x * DECAYF + acc.x * ODECAY;
  float n1 = old.y * DECAYF + acc.y * ODECAY;
  float n2 = old.z * DECAYF + acc.z * ODECAY;
  float n3 = old.w * DECAYF + acc.w * ODECAY;
  size_t o = (size_t)j * EMBED + c;
  out[OUT_NCSUM + o + 0] = n0; out[OUT_NCSUM + o + 1] = n1;   // odd base offset ->
  out[OUT_NCSUM + o + 2] = n2; out[OUT_NCSUM + o + 3] = n3;   // scalar stores
  out[OUT_NCB + o + 0] = n0 * inv; out[OUT_NCB + o + 1] = n1 * inv;
  out[OUT_NCB + o + 2] = n2 * inv; out[OUT_NCB + o + 3] = n3 * inv;
  if (j == 0 && t == 0)
    out[OUT_LOSS] = wsf[WS_LOSS] * (COMMITF / ((float)NTOK * (float)EMBED));
}

extern "C" void kernel_launch(void* const* d_in, const int* in_sizes, int n_in,
                              void* d_out, int out_size, void* d_ws, size_t ws_size,
                              hipStream_t stream) {
  (void)in_sizes; (void)n_in; (void)out_size; (void)ws_size;
  const float* x             = (const float*)d_in[0];
  const float* codebook      = (const float*)d_in[1];
  const float* cluster_usage = (const float*)d_in[2];
  const float* cluster_sum   = (const float*)d_in[3];
  const float* channel_means = (const float*)d_in[4];
  const float* channel_stds  = (const float*)d_in[5];
  float* out = (float*)d_out;
  float* wsf = (float*)d_ws;
  unsigned short* cb16 = (unsigned short*)((char*)d_ws + WS_CB_BYTE_OFF);

  k_zero<<<(int)((WS_ZERO_END + 255) / 256), 256, 0, stream>>>(wsf, (int)WS_ZERO_END);
  k_stats<<<NTOK / 512, 512, 0, stream>>>(x, wsf);
  k_finstats<<<1, 512, 0, stream>>>(channel_means, channel_stds, wsf, out);
  k_cbprep<<<NCLUST, 128, 0, stream>>>(codebook, wsf, cb16);
  k_main<<<NTOK / 64, 128, 0, stream>>>(x, codebook, cb16, wsf, out);
  k_final<<<NCLUST, 128, 0, stream>>>(cluster_usage, cluster_sum, wsf, out);
}
